// EncoderLayer_70214125355185
// MI455X (gfx1250) — matrix operand order repaired and checked by reading, not compiled, not run on hardware
//
#include <hip/hip_runtime.h>
#include <hip/hip_bf16.h>
#include <math.h>

// ---------------------------------------------------------------------------
// Transformer encoder layer (multiplicative attention variant) for gfx1250.
// fp32 reference emulated with bf16x2-split WMMA (hi*hi + hi*lo + lo*hi),
// fp32 accumulation -> ~1e-5 relative error at bf16 matrix-pipe speed.
// GEMM B tiles are staged into LDS with async global->LDS copies (double
// buffered, ASYNCcnt-tracked).
// ---------------------------------------------------------------------------

typedef __bf16 bf16_t;
typedef __attribute__((ext_vector_type(16))) __bf16 v16bf;
typedef __attribute__((ext_vector_type(8)))  __bf16 v8bf;
typedef __attribute__((ext_vector_type(8)))  float  v8f;

// Problem dims (fixed by the reference)
constexpr int Bb   = 4;
constexpr int Ss   = 2048;
constexpr int HIDc = 512;
constexpr int Hh   = 8;
constexpr int Dd   = 64;      // HID / H
constexpr int PFc  = 2048;
constexpr int MR   = Bb * Ss;     // 8192 token rows
constexpr int BH   = Bb * Hh;     // 32 (batch, head) pairs

#define DEVFN static __device__ __forceinline__

#if defined(__has_builtin)
#if __has_builtin(__builtin_amdgcn_global_load_async_to_lds_b128)
#define HAVE_ASYNC_LDS 1
#endif
#if __has_builtin(__builtin_amdgcn_s_wait_asynccnt)
#define HAVE_WAIT_ASYNC 1
#endif
#endif

// Pointer types matching the async-to-LDS builtin signature exactly:
// param0: <4 x i32>* in global AS(1); param1: <4 x i32>* in LDS AS(3).
typedef int v4i_ __attribute__((vector_size(16)));
typedef __attribute__((address_space(1))) v4i_ g1_v4i;
typedef __attribute__((address_space(3))) v4i_ l3_v4i;

// Low 32 bits of a generic shared pointer are the LDS byte offset
// (flat-aperture rule), so an integer cast yields a valid AS(3) pointer.
DEVFN l3_v4i* to_lds(void* p) { return (l3_v4i*)(uint32_t)(uintptr_t)p; }
DEVFN g1_v4i* to_glb(const void* p) { return (g1_v4i*)(uintptr_t)p; }

DEVFN void wait_async0() {
#if defined(HAVE_WAIT_ASYNC)
  __builtin_amdgcn_s_wait_asynccnt(0);
#elif defined(HAVE_ASYNC_LDS)
  asm volatile("s_wait_asynccnt 0" ::: "memory");
#endif
}

DEVFN v8f vzero8() { v8f c = {0.f,0.f,0.f,0.f,0.f,0.f,0.f,0.f}; return c; }

DEVFN v16bf cat8(v8bf a, v8bf b) {
  return __builtin_shufflevector(a, b, 0,1,2,3,4,5,6,7,8,9,10,11,12,13,14,15);
}

// B fragment (32x16 tile, K-major storage): lane L holds column N = L%16,
// K = k0+{0..7, 16..23} (lanes 0-15) / k0+{8..15, 24..31} (lanes 16-31).
// Caller passes &B[k0*ldb + n0 + lane%16].
DEVFN v16bf load_b_frag(const bf16_t* p, int ldb, int lhalf) {
  v16bf r;
  int b8 = lhalf * 8;
#pragma unroll
  for (int i = 0; i < 16; ++i) {
    int k = (i < 8) ? (b8 + i) : (16 + b8 + (i - 8));
    r[i] = p[(size_t)k * ldb];
  }
  return r;
}

// A fragment (16x32 tile, row-major): lane L holds row M = L%16.
// lanes 0-15: K = k0+{0..7, 16..23}; lanes 16-31: K = k0+{8..15, 24..31}.
// Caller passes &A[m_row*lda + k0]; lhalf = lane>>4.
DEVFN v16bf load_a_frag(const bf16_t* rowk0, int lhalf) {
  int b8 = lhalf * 8;
  v8bf c0 = *(const v8bf*)(rowk0 + b8);
  v8bf c1 = *(const v8bf*)(rowk0 + 16 + b8);
  return cat8(c0, c1);
}

DEVFN v8f wmma_bf16(v16bf a, v16bf b, v8f c) {
  return __builtin_amdgcn_wmma_f32_16x16x32_bf16(false, a, false, b,
                                                 (short)0, c, false, false);
}

// Split-WMMA: C += (Ah+Al)*(Bh+Bl), dropping lo*lo.
DEVFN v8f wmma_split(v16bf ah, v16bf al, v16bf bh, v16bf bl, v8f c) {
  c = wmma_bf16(ah, bh, c);
  c = wmma_bf16(ah, bl, c);
  c = wmma_bf16(al, bh, c);
  return c;
}

DEVFN void split_store(bf16_t* h, bf16_t* l, size_t idx, float v) {
  bf16_t hi = (bf16_t)v;
  h[idx] = hi;
  l[idx] = (bf16_t)(v - (float)hi);
}

// ---------------------------------------------------------------------------
// Prep kernels: fp32 -> bf16 hi/lo (optionally transposed for weights)
// ---------------------------------------------------------------------------
__global__ void k_split(const float* __restrict__ x, bf16_t* __restrict__ h,
                        bf16_t* __restrict__ l, int n) {
  int g = blockIdx.x * blockDim.x + threadIdx.x;
  if (g < n) split_store(h, l, (size_t)g, x[g]);
}

// W[o][i] (row-major O x I) -> T[i][o] hi/lo (K-major for WMMA B operand)
__global__ void k_tsplit(const float* __restrict__ W, bf16_t* __restrict__ Th,
                         bf16_t* __restrict__ Tl, int O, int I) {
  int g = blockIdx.x * blockDim.x + threadIdx.x;
  if (g < O * I) {
    int o = g / I, i = g - o * I;
    split_store(Th, Tl, (size_t)i * O + o, W[g]);
  }
}

// ---------------------------------------------------------------------------
// Generic bf16x2-split GEMM:  C[m][n] = sum_k A[m][k]*B[k][n] + bias[n]
// block = 256 threads (8 waves); wave tile 32x64; block tile 256x64.
// grid = (N/64, M/256).  Requires M%256==0, N%64==0, K%32==0.
// B tiles (shared by all 8 waves) are staged to LDS, double-buffered, via
// async global->LDS copies when available.
// ---------------------------------------------------------------------------
enum { EP_F32 = 0, EP_SPLIT = 1, EP_RELU_SPLIT = 2, EP_QV = 3, EP_KT = 4 };

__global__ __launch_bounds__(256)
void k_gemm(const bf16_t* __restrict__ Ah, const bf16_t* __restrict__ Al,
            const bf16_t* __restrict__ Bh, const bf16_t* __restrict__ Bl,
            const float* __restrict__ bias,
            float* __restrict__ outf, bf16_t* __restrict__ outh,
            bf16_t* __restrict__ outl,
            int M, int N, int K, int mode) {
  __shared__ __align__(16) bf16_t BsH[2][32][64];
  __shared__ __align__(16) bf16_t BsL[2][32][64];

  const int tid   = threadIdx.x;
  const int wv    = tid >> 5;
  const int lane  = tid & 31;
  const int lhalf = lane >> 4;
  const int l16   = lane & 15;

  const int m0 = blockIdx.y * 256 + wv * 32;
  const int n0 = blockIdx.x * 64;

  // Staging assignment: 256 threads x (one 16B chunk each, hi+lo) covers the
  // 32x64 tile (row = tid/8, 8 bf16 starting at col 8*(tid%8)).
  const int srow = tid >> 3;
  const int scol = (tid & 7) * 8;

  auto stage = [&](int buf, int k0) {
    const bf16_t* gh = Bh + (size_t)(k0 + srow) * N + n0 + scol;
    const bf16_t* gl = Bl + (size_t)(k0 + srow) * N + n0 + scol;
    bf16_t* dh = &BsH[buf][srow][scol];
    bf16_t* dl = &BsL[buf][srow][scol];
#if defined(HAVE_ASYNC_LDS)
    __builtin_amdgcn_global_load_async_to_lds_b128(to_glb(gh), to_lds(dh), 0, 0);
    __builtin_amdgcn_global_load_async_to_lds_b128(to_glb(gl), to_lds(dl), 0, 0);
#else
    *(v8bf*)dh = *(const v8bf*)gh;
    *(v8bf*)dl = *(const v8bf*)gl;
#endif
  };

  v8f acc[2][4];
#pragma unroll
  for (int ms = 0; ms < 2; ++ms)
#pragma unroll
    for (int nt = 0; nt < 4; ++nt) acc[ms][nt] = vzero8();

  stage(0, 0);
  int buf = 0;

  for (int k0 = 0; k0 < K; k0 += 32) {
    wait_async0();        // this wave's staging of `buf` complete
    __syncthreads();      // tile visible to all waves; prev readers done
    if (k0 + 32 < K) {
      stage(buf ^ 1, k0 + 32);
      // prefetch next A chunk for this wave's rows
      __builtin_prefetch(Ah + (size_t)(m0 + l16) * K + k0 + 32, 0, 1);
      __builtin_prefetch(Al + (size_t)(m0 + l16) * K + k0 + 32, 0, 1);
    }

    v16bf a_h[2], a_l[2];
#pragma unroll
    for (int ms = 0; ms < 2; ++ms) {
      size_t row = (size_t)(m0 + ms * 16 + l16);
      a_h[ms] = load_a_frag(Ah + row * K + k0, lhalf);
      a_l[ms] = load_a_frag(Al + row * K + k0, lhalf);
    }
#pragma unroll
    for (int nt = 0; nt < 4; ++nt) {
      v16bf b_h = load_b_frag(&BsH[buf][0][nt * 16 + l16], 64, lhalf);
      v16bf b_l = load_b_frag(&BsL[buf][0][nt * 16 + l16], 64, lhalf);
#pragma unroll
      for (int ms = 0; ms < 2; ++ms)
        acc[ms][nt] = wmma_split(a_h[ms], a_l[ms], b_h, b_l, acc[ms][nt]);
    }
    __syncthreads();      // all waves done reading `buf` before restage
    buf ^= 1;
  }

  // Epilogue. C fragment: VGPR r -> M = r + 8*lhalf, N = lane%16.
#pragma unroll
  for (int ms = 0; ms < 2; ++ms) {
#pragma unroll
    for (int nt = 0; nt < 4; ++nt) {
      v8f c = acc[ms][nt];
#pragma unroll
      for (int r = 0; r < 8; ++r) {
        int m = m0 + ms * 16 + r + lhalf * 8;
        int n = n0 + nt * 16 + l16;
        float v = c[r] + bias[n];
        if (mode == EP_F32) {
          outf[(size_t)m * N + n] = v;
        } else if (mode == EP_SPLIT) {
          split_store(outh, outl, (size_t)m * N + n, v);
        } else if (mode == EP_RELU_SPLIT) {
          split_store(outh, outl, (size_t)m * N + n, fmaxf(v, 0.f));
        } else if (mode == EP_QV) {
          int b = m >> 11, s = m & (Ss - 1);        // m = b*S + s
          int h = n >> 6, d = n & (Dd - 1);         // n = h*D + d
          size_t idx = (((size_t)(b * Hh + h)) * Ss + s) * Dd + d;
          split_store(outh, outl, idx, v);
        } else { // EP_KT: K stored transposed [B,H,D,S]
          int b = m >> 11, s = m & (Ss - 1);
          int h = n >> 6, d = n & (Dd - 1);
          size_t idx = (((size_t)(b * Hh + h)) * Dd + d) * Ss + s;
          split_store(outh, outl, idx, v);
        }
      }
    }
  }
}

// ---------------------------------------------------------------------------
// Flash attention (multiplicative, no 1/sqrt(d), key mask, ctx -> bf16 hi/lo
// in flat [B,H,S,D] order -- the reference's reshape bug makes this directly
// the Wo GEMM input).  One wave per 16-query tile; keys streamed 32 at a time.
// ---------------------------------------------------------------------------
__global__ __launch_bounds__(256)
void k_flash(const bf16_t* __restrict__ Qmh, const bf16_t* __restrict__ Qml,
             const bf16_t* __restrict__ Kth, const bf16_t* __restrict__ Ktl,
             const bf16_t* __restrict__ Vh,  const bf16_t* __restrict__ Vl,
             const int* __restrict__ mask,
             bf16_t* __restrict__ ctxh, bf16_t* __restrict__ ctxl) {
  __shared__ float Plds[8 * 16 * 32];        // per-wave 16x32 prob tile

  const int tid   = threadIdx.x;
  const int wv    = tid >> 5;
  const int lane  = tid & 31;
  const int lhalf = lane >> 4;
  const int l16   = lane & 15;

  const int gw = blockIdx.x * 8 + wv;        // global wave id
  const int bh = gw / (Ss / 16);             // (batch, head)
  const int q0 = (gw % (Ss / 16)) * 16;      // query tile base
  const int b_ = bh >> 3;                    // batch (H = 8)

  float* pls = Plds + wv * (16 * 32);

  // Qm A-fragments for both 32-wide d chunks (row = q0 + lane%16)
  v16bf aQh[2], aQl[2];
  {
    size_t row = (size_t)bh * Ss + q0 + l16;
#pragma unroll
    for (int dk = 0; dk < 2; ++dk) {
      aQh[dk] = load_a_frag(Qmh + row * Dd + dk * 32, lhalf);
      aQl[dk] = load_a_frag(Qml + row * Dd + dk * 32, lhalf);
    }
  }

  v8f accf[4];
#pragma unroll
  for (int dt = 0; dt < 4; ++dt) accf[dt] = vzero8();
  float mrow[8], lrow[8];
#pragma unroll
  for (int r = 0; r < 8; ++r) { mrow[r] = -__builtin_inff(); lrow[r] = 0.f; }

  for (int kc = 0; kc < Ss; kc += 32) {
    // ---- energy tiles: e[j] = Qm (16xD) @ Kt (D x 16) for keys kc+16j ----
    v8f e[2] = {vzero8(), vzero8()};
#pragma unroll
    for (int j = 0; j < 2; ++j) {
      int n0 = kc + j * 16;
#pragma unroll
      for (int dk = 0; dk < 2; ++dk) {
        size_t boff = ((size_t)bh * Dd + dk * 32) * Ss + n0 + l16;
        v16bf bKh = load_b_frag(Kth + boff, Ss, lhalf);
        v16bf bKl = load_b_frag(Ktl + boff, Ss, lhalf);
        e[j] = wmma_split(aQh[dk], aQl[dk], bKh, bKl, e[j]);
      }
    }
    // ---- key mask (per N = lane%16, uniform over the 8 fragment rows) ----
#pragma unroll
    for (int j = 0; j < 2; ++j) {
      int mk = mask[b_ * Ss + kc + j * 16 + l16];
      if (mk == 0) {
#pragma unroll
        for (int r = 0; r < 8; ++r) e[j][r] = -1e10f;
      }
    }
    // ---- online softmax: row max / rescale / exp / row sum ----
    float mnew[8], scl[8];
#pragma unroll
    for (int r = 0; r < 8; ++r) {
      float em = fmaxf(e[0][r], e[1][r]);
#pragma unroll
      for (int off = 8; off >= 1; off >>= 1)
        em = fmaxf(em, __shfl_xor(em, off, 32));
      mnew[r] = fmaxf(mrow[r], em);
      scl[r]  = __expf(mrow[r] - mnew[r]);
      mrow[r] = mnew[r];
    }
#pragma unroll
    for (int j = 0; j < 2; ++j)
#pragma unroll
      for (int r = 0; r < 8; ++r) e[j][r] = __expf(e[j][r] - mnew[r]);
#pragma unroll
    for (int r = 0; r < 8; ++r) {
      float ps = e[0][r] + e[1][r];
#pragma unroll
      for (int off = 8; off >= 1; off >>= 1) ps += __shfl_xor(ps, off, 32);
      lrow[r] = lrow[r] * scl[r] + ps;
    }
#pragma unroll
    for (int dt = 0; dt < 4; ++dt)
#pragma unroll
      for (int r = 0; r < 8; ++r) accf[dt][r] *= scl[r];

    // ---- C-layout -> A-layout remap of P through per-wave LDS tile ----
#pragma unroll
    for (int j = 0; j < 2; ++j)
#pragma unroll
      for (int r = 0; r < 8; ++r)
        pls[(r + 8 * lhalf) * 32 + j * 16 + l16] = e[j][r];
    asm volatile("s_wait_dscnt 0" ::: "memory");

    v16bf ph, pl;
    {
      const float* prow = pls + l16 * 32;
      int b8 = lhalf * 8;
#pragma unroll
      for (int i = 0; i < 16; ++i) {
        int k = (i < 8) ? (b8 + i) : (16 + b8 + (i - 8));
        float v = prow[k];
        bf16_t hi = (bf16_t)v;
        ph[i] = hi;
        pl[i] = (bf16_t)(v - (float)hi);
      }
    }
    asm volatile("s_wait_dscnt 0" ::: "memory");

    // ---- accf += P (16x32) @ V (32 x 16) per d tile ----
#pragma unroll
    for (int dt = 0; dt < 4; ++dt) {
      size_t voff = ((size_t)bh * Ss + kc) * Dd + dt * 16 + l16;
      v16bf bVh = load_b_frag(Vh + voff, Dd, lhalf);
      v16bf bVl = load_b_frag(Vl + voff, Dd, lhalf);
      accf[dt] = wmma_split(ph, pl, bVh, bVl, accf[dt]);
    }
  }

  // ---- normalize and store ctx (flat [B,H,S,D]) as bf16 hi/lo ----
#pragma unroll
  for (int dt = 0; dt < 4; ++dt) {
#pragma unroll
    for (int r = 0; r < 8; ++r) {
      float v = accf[dt][r] / lrow[r];
      size_t idx = ((size_t)bh * Ss + q0 + r + 8 * lhalf) * Dd + dt * 16 + l16;
      split_store(ctxh, ctxl, idx, v);
    }
  }
}

// ---------------------------------------------------------------------------
// Residual + LayerNorm, one wave per 512-wide row (lane handles 16 cols).
// Optionally re-splits the output to bf16 hi/lo for the next GEMM.
// ---------------------------------------------------------------------------
__global__ __launch_bounds__(256)
void k_add_ln(const float* __restrict__ X, const float* __restrict__ Y,
              const float* __restrict__ g, const float* __restrict__ bta,
              float* __restrict__ outf, bf16_t* __restrict__ outh,
              bf16_t* __restrict__ outl, int rows) {
  const int wv = threadIdx.x >> 5, lane = threadIdx.x & 31;
  const int row = blockIdx.x * 8 + wv;
  if (row >= rows) return;
  const size_t base = (size_t)row * HIDc;
  const int c0 = lane * 16;

  float v[16], s = 0.f;
#pragma unroll
  for (int i = 0; i < 16; ++i) { v[i] = X[base + c0 + i] + Y[base + c0 + i]; s += v[i]; }
#pragma unroll
  for (int off = 16; off >= 1; off >>= 1) s += __shfl_xor(s, off, 32);
  const float mean = s * (1.f / HIDc);

  float q = 0.f;
#pragma unroll
  for (int i = 0; i < 16; ++i) { float d = v[i] - mean; q += d * d; }
#pragma unroll
  for (int off = 16; off >= 1; off >>= 1) q += __shfl_xor(q, off, 32);
  const float rstd = rsqrtf(q * (1.f / HIDc) + 1e-5f);

#pragma unroll
  for (int i = 0; i < 16; ++i) {
    float o = (v[i] - mean) * rstd * g[c0 + i] + bta[c0 + i];
    size_t idx = base + c0 + i;
    if (outf) outf[idx] = o;
    if (outh) split_store(outh, outl, idx, o);
  }
}

// ---------------------------------------------------------------------------
// Host launcher
// ---------------------------------------------------------------------------
extern "C" void kernel_launch(void* const* d_in, const int* in_sizes, int n_in,
                              void* d_out, int out_size, void* d_ws, size_t ws_size,
                              hipStream_t stream) {
  (void)in_sizes; (void)n_in; (void)out_size; (void)ws_size;
  const float* src   = (const float*)d_in[0];
  const int*   smask = (const int*)  d_in[1];
  const float* Wq = (const float*)d_in[2];  const float* bq = (const float*)d_in[3];
  const float* Wk = (const float*)d_in[4];  const float* bk = (const float*)d_in[5];
  const float* Wv = (const float*)d_in[6];  const float* bv = (const float*)d_in[7];
  const float* Wm = (const float*)d_in[8];  const float* bm = (const float*)d_in[9];
  const float* Wo = (const float*)d_in[10]; const float* bo = (const float*)d_in[11];
  const float* g1 = (const float*)d_in[12]; const float* be1 = (const float*)d_in[13];
  const float* W1 = (const float*)d_in[14]; const float* b1 = (const float*)d_in[15];
  const float* W2 = (const float*)d_in[16]; const float* b2 = (const float*)d_in[17];
  const float* g2 = (const float*)d_in[18]; const float* be2 = (const float*)d_in[19];
  float* out = (float*)d_out;

  // Workspace bump allocator (256B aligned slices)
  size_t ws_off = 0;
  auto alloc = [&](size_t nbytes) -> char* {
    char* p = (char*)d_ws + ws_off;
    ws_off += (nbytes + 255) & ~(size_t)255;
    return p;
  };
  const size_t NT = (size_t)MR * HIDc;            // 8192*512 token elements
  auto bfpair = [&](bf16_t*& h, bf16_t*& l, size_t n) {
    h = (bf16_t*)alloc(n * 2); l = (bf16_t*)alloc(n * 2);
  };

  bf16_t *srch, *srcl;               bfpair(srch, srcl, NT);
  bf16_t *WqTh, *WqTl;               bfpair(WqTh, WqTl, (size_t)HIDc * HIDc);
  bf16_t *WkTh, *WkTl;               bfpair(WkTh, WkTl, (size_t)HIDc * HIDc);
  bf16_t *WvTh, *WvTl;               bfpair(WvTh, WvTl, (size_t)HIDc * HIDc);
  bf16_t *WmTh, *WmTl;               bfpair(WmTh, WmTl, (size_t)Dd * Dd);
  bf16_t *WoTh, *WoTl;               bfpair(WoTh, WoTl, (size_t)HIDc * HIDc);
  bf16_t *W1Th, *W1Tl;               bfpair(W1Th, W1Tl, (size_t)HIDc * PFc);
  bf16_t *W2Th, *W2Tl;               bfpair(W2Th, W2Tl, (size_t)PFc * HIDc);
  bf16_t *Qh, *Ql;                   bfpair(Qh, Ql, NT);      // [B,H,S,D]
  bf16_t *Kth, *Ktl;                 bfpair(Kth, Ktl, NT);    // [B,H,D,S]
  bf16_t *Vh, *Vl;                   bfpair(Vh, Vl, NT);      // [B,H,S,D]
  bf16_t *Qmh, *Qml;                 bfpair(Qmh, Qml, NT);    // [B,H,S,D]
  bf16_t *ctxh, *ctxl;               bfpair(ctxh, ctxl, NT);  // flat [B,H,S,D]
  float* attnout = (float*)alloc(NT * 4);
  float* src1f   = (float*)alloc(NT * 4);
  bf16_t *s1h, *s1l;                 bfpair(s1h, s1l, NT);
  bf16_t *f1h, *f1l;                 bfpair(f1h, f1l, (size_t)MR * PFc);
  float* ff2     = (float*)alloc(NT * 4);

  const dim3 blk(256);
  // ---- prep: splits / transposed weight splits ----
  k_split<<<dim3((unsigned)(NT / 256)), blk, 0, stream>>>(src, srch, srcl, (int)NT);
  auto tsp = [&](const float* W, bf16_t* Th, bf16_t* Tl, int O, int I) {
    int n = O * I;
    k_tsplit<<<dim3((unsigned)((n + 255) / 256)), blk, 0, stream>>>(W, Th, Tl, O, I);
  };
  tsp(Wq, WqTh, WqTl, HIDc, HIDc);
  tsp(Wk, WkTh, WkTl, HIDc, HIDc);
  tsp(Wv, WvTh, WvTl, HIDc, HIDc);
  tsp(Wm, WmTh, WmTl, Dd, Dd);
  tsp(Wo, WoTh, WoTl, HIDc, HIDc);
  tsp(W1, W1Th, W1Tl, PFc, HIDc);    // -> [HID][PF]
  tsp(W2, W2Th, W2Tl, HIDc, PFc);    // -> [PF][HID]

  auto gemm = [&](const bf16_t* Ah, const bf16_t* Al, const bf16_t* Bh,
                  const bf16_t* Bl, const float* bias, float* of, bf16_t* oh,
                  bf16_t* ol, int M, int N, int K, int mode) {
    dim3 grid((unsigned)(N / 64), (unsigned)(M / 256));
    k_gemm<<<grid, blk, 0, stream>>>(Ah, Al, Bh, Bl, bias, of, oh, ol, M, N, K, mode);
  };

  // ---- QKV projections ----
  gemm(srch, srcl, WqTh, WqTl, bq, nullptr, Qh, Ql,  MR, HIDc, HIDc, EP_QV);
  gemm(srch, srcl, WkTh, WkTl, bk, nullptr, Kth, Ktl, MR, HIDc, HIDc, EP_KT);
  gemm(srch, srcl, WvTh, WvTl, bv, nullptr, Vh, Vl,  MR, HIDc, HIDc, EP_QV);
  // ---- Qm = Q @ Wm^T + bm (per head; rows = B*H*S) ----
  gemm(Qh, Ql, WmTh, WmTl, bm, nullptr, Qmh, Qml, MR * Hh, Dd, Dd, EP_SPLIT);
  // ---- flash attention ----
  k_flash<<<dim3((unsigned)(BH * (Ss / 16) / 8)), blk, 0, stream>>>(
      Qmh, Qml, Kth, Ktl, Vh, Vl, smask, ctxh, ctxl);
  // ---- output projection (ctx flat == reference's buggy reshape) ----
  gemm(ctxh, ctxl, WoTh, WoTl, bo, attnout, nullptr, nullptr, MR, HIDc, HIDc, EP_F32);
  // ---- residual + LN1 (also re-split for FFN) ----
  k_add_ln<<<dim3((unsigned)(MR / 8)), blk, 0, stream>>>(src, attnout, g1, be1,
                                                         src1f, s1h, s1l, MR);
  // ---- FFN ----
  gemm(s1h, s1l, W1Th, W1Tl, b1, nullptr, f1h, f1l, MR, PFc, HIDc, EP_RELU_SPLIT);
  gemm(f1h, f1l, W2Th, W2Tl, b2, ff2, nullptr, nullptr, MR, HIDc, PFc, EP_F32);
  // ---- residual + LN2 -> final output ----
  k_add_ln<<<dim3((unsigned)(MR / 8)), blk, 0, stream>>>(src1f, ff2, g2, be2,
                                                         out, nullptr, nullptr, MR);
}
